// NeuralMemory_16183436771555
// MI455X (gfx1250) — matrix operand order repaired and checked by reading, not compiled, not run on hardware
//
#include <hip/hip_runtime.h>
#include <hip/hip_bf16.h>

#define B_   2
#define N_   2048
#define DIM_ 512
#define H_   4
#define DH_  128
#define HID_ 512
#define CH_  64
#define NC_  32
#define WEL_ (DH_ * HID_)
#define EPS_ 1e-6f

typedef __attribute__((ext_vector_type(16))) __bf16 v16bf;
typedef __attribute__((ext_vector_type(8)))  float  v8f;

// ---------------- WMMA helpers (CDNA5 wave32, 16x16x32 bf16 -> f32) ----------------

__device__ inline v8f wmma_bf16(v16bf a, v16bf b, v8f c) {
  return __builtin_amdgcn_wmma_f32_16x16x32_bf16(false, a, false, b, (short)0, c, false, false);
}

__device__ inline v8f zero8() {
  v8f z;
#pragma unroll
  for (int i = 0; i < 8; ++i) z[i] = 0.f;
  return z;
}

// A fragment (16x32, MxK). Lane L<16: M=L, K=0..15. Lane 16+L: M=L, K=16..31.
// Element (m, k) at p0[m*rs + k*cs]. cs==1 -> one contiguous 16-element run per lane.
template <typename T>
__device__ inline v16bf load_afrag(const T* p0, int rs, int cs) {
  int lane = threadIdx.x & 31;
  int half = lane >> 4, m = lane & 15;
  const T* p = p0 + m * rs;
  v16bf a;
#pragma unroll
  for (int e = 0; e < 8; ++e) {
    a[e]     = (__bf16)(float)p[(half * 16 + e) * cs];
    a[e + 8] = (__bf16)(float)p[(half * 16 + 8 + e) * cs];
  }
  return a;
}

// B fragment (32x16, KxN). Lane L<16: N=L, K=0..15. Lane 16+L: N=L, K=16..31.
// Element (k, n) at p0[k*rs + n*cs]. rs==1 -> one contiguous 16-element run per lane.
template <typename T>
__device__ inline v16bf load_bfrag(const T* p0, int rs, int cs) {
  int lane = threadIdx.x & 31;
  int half = lane >> 4, n = lane & 15;
  const T* p = p0 + (half * 16) * rs + n * cs;
  v16bf b;
#pragma unroll
  for (int e = 0; e < 16; ++e) b[e] = (__bf16)(float)p[e * rs];
  return b;
}

// C/D fragment store: VGPR r of lane L -> (M = r + 8*(L/16), N = L%16); element (m,n) at p0[m*rs + n*cs]
template <typename T>
__device__ inline void store_cfrag(T* p0, int rs, int cs, v8f c) {
  int lane = threadIdx.x & 31;
  int half = lane >> 4, n = lane & 15;
#pragma unroll
  for (int r = 0; r < 8; ++r) p0[(r + 8 * half) * rs + n * cs] = (T)c[r];
}

__device__ inline float geluf(float x) {
  return 0.5f * x * (1.f + erff(x * 0.70710678118f));
}
__device__ inline float gelu_grad(float x) {
  float cdf = 0.5f * (1.f + erff(x * 0.70710678118f));
  float pdf = 0.3989422804f * __expf(-0.5f * x * x);
  return cdf + x * pdf;
}
__device__ inline float sigmoidf_(float x) { return 1.f / (1.f + __expf(-x)); }

// ---------------- 1) RMS-norm ----------------

__global__ void rms_kernel(const float* __restrict__ x, const float* __restrict__ g_ret,
                           const float* __restrict__ g_sto, float* __restrict__ xs,
                           float* __restrict__ xr) {
  __shared__ float red[256];
  int row = blockIdx.x;
  int tid = threadIdx.x;
  const float* xp = x + (size_t)row * DIM_;
  float s = 0.f;
  for (int d = tid; d < DIM_; d += 256) { float v = xp[d]; s += v * v; }
  red[tid] = s; __syncthreads();
  for (int o = 128; o > 0; o >>= 1) {
    if (tid < o) red[tid] += red[tid + o];
    __syncthreads();
  }
  float rstd = rsqrtf(red[0] / (float)DIM_ + EPS_);
  for (int d = tid; d < DIM_; d += 256) {
    float v = xp[d] * rstd;
    xs[(size_t)row * DIM_ + d] = v * g_sto[d];
    xr[(size_t)row * DIM_ + d] = v * g_ret[d];
  }
}

// ---------------- 2) Generic WMMA GEMM: C[M,N] = A[M,K] @ B[K,N] (+bias) ----------------
// 64x64 block / 8 waves, float4 global staging, B staged K-contiguous (transposed) in LDS.

__global__ void gemm_kernel(const float* __restrict__ A, const float* __restrict__ B,
                            const float* __restrict__ bias, float* __restrict__ C,
                            int M, int N, int K) {
  __shared__ __bf16 As[64 * 32];   // [m][k]
  __shared__ __bf16 Bst[64 * 32];  // [n][k]  (K unit-stride for B fragments)
  int tid = threadIdx.x;
  int wave = tid >> 5;
  int lane = tid & 31;
  int col = lane & 15;
  int m0 = blockIdx.y * 64, n0 = blockIdx.x * 64;

  v8f acc0 = zero8(), acc1 = zero8();
  for (int kk = 0; kk < K; kk += 32) {
    for (int i = tid; i < 64 * 8; i += 256) {
      int r = i >> 3, c4 = (i & 7) * 4;
      float4 f = *(const float4*)(A + (size_t)(m0 + r) * K + kk + c4);
      As[r * 32 + c4 + 0] = (__bf16)f.x;
      As[r * 32 + c4 + 1] = (__bf16)f.y;
      As[r * 32 + c4 + 2] = (__bf16)f.z;
      As[r * 32 + c4 + 3] = (__bf16)f.w;
    }
    for (int i = tid; i < 32 * 16; i += 256) {
      int r = i >> 4, c4 = (i & 15) * 4;
      float4 f = *(const float4*)(B + (size_t)(kk + r) * N + n0 + c4);
      Bst[(c4 + 0) * 32 + r] = (__bf16)f.x;
      Bst[(c4 + 1) * 32 + r] = (__bf16)f.y;
      Bst[(c4 + 2) * 32 + r] = (__bf16)f.z;
      Bst[(c4 + 3) * 32 + r] = (__bf16)f.w;
    }
    if (kk + 32 < K) {  // prefetch next K tile (global_prefetch_b8)
      __builtin_prefetch(A + (size_t)(m0 + (tid & 63)) * K + kk + 32, 0, 1);
      __builtin_prefetch(B + (size_t)(kk + 32 + (tid & 31)) * N + n0, 0, 1);
    }
    __syncthreads();
#pragma unroll
    for (int t = 0; t < 2; ++t) {
      int tile = wave + t * 8;
      int mt = tile >> 2, nt = tile & 3;
      v16bf a = load_afrag(As + mt * 16 * 32, 32, 1);
      v16bf b = load_bfrag(Bst + nt * 16 * 32, 1, 32);
      if (t == 0) acc0 = wmma_bf16(a, b, acc0);
      else        acc1 = wmma_bf16(a, b, acc1);
    }
    __syncthreads();
  }
#pragma unroll
  for (int t = 0; t < 2; ++t) {
    int tile = wave + t * 8;
    int mt = tile >> 2, nt = tile & 3;
    v8f c = (t == 0) ? acc0 : acc1;
    if (bias) {
      float bn = bias[n0 + nt * 16 + col];
#pragma unroll
      for (int r = 0; r < 8; ++r) c[r] += bn;
    }
    store_cfrag(C + (size_t)(m0 + mt * 16) * N + n0 + nt * 16, N, 1, c);
  }
}

// ---------------- 3) per-(token,head) L2-normalize rows of 128 ----------------

__global__ void l2n_kernel(float* __restrict__ t) {
  __shared__ float red[128];
  int tid = threadIdx.x;
  float* p = t + (size_t)blockIdx.x * DH_;
  float v = p[tid];
  red[tid] = v * v; __syncthreads();
  for (int o = 64; o > 0; o >>= 1) {
    if (tid < o) red[tid] += red[tid + o];
    __syncthreads();
  }
  p[tid] = v * rsqrtf(red[0] + 1e-12f);
}

// ---------------- 4) lr (from xs) and gate (from xr) ----------------

__global__ void aux_kernel(const float* __restrict__ xs, const float* __restrict__ xr,
                           const float* __restrict__ W_lr, const float* __restrict__ b_lr,
                           const float* __restrict__ W_gate, const float* __restrict__ b_gate,
                           float* __restrict__ lr, float* __restrict__ gate) {
  __shared__ float red[4][256];
  int tid = threadIdx.x;
  int row = blockIdx.x;                 // b*N + n
  int bidx = row / N_, n = row % N_;
  const float* xsp = xs + (size_t)row * DIM_;
  const float* xrp = xr + (size_t)row * DIM_;

  float pl[4] = {0, 0, 0, 0};
  for (int d = tid; d < DIM_; d += 256) {
    float v = xsp[d];
#pragma unroll
    for (int h = 0; h < H_; ++h) pl[h] += v * W_lr[d * H_ + h];
  }
#pragma unroll
  for (int h = 0; h < H_; ++h) red[h][tid] = pl[h];
  __syncthreads();
  for (int o = 128; o > 0; o >>= 1) {
    if (tid < o) for (int h = 0; h < H_; ++h) red[h][tid] += red[h][tid + o];
    __syncthreads();
  }
  if (tid < H_) lr[((size_t)(bidx * H_ + tid)) * N_ + n] =
      sigmoidf_(red[tid][0] + b_lr[tid]) * 0.01f;
  __syncthreads();

  float pg[4] = {0, 0, 0, 0};
  for (int d = tid; d < DIM_; d += 256) {
    float v = xrp[d];
#pragma unroll
    for (int h = 0; h < H_; ++h) pg[h] += v * W_gate[d * H_ + h];
  }
#pragma unroll
  for (int h = 0; h < H_; ++h) red[h][tid] = pg[h];
  __syncthreads();
  for (int o = 128; o > 0; o >>= 1) {
    if (tid < o) for (int h = 0; h < H_; ++h) red[h][tid] += red[h][tid + o];
    __syncthreads();
  }
  if (tid < H_) gate[((size_t)(bidx * H_ + tid)) * N_ + n] =
      sigmoidf_(red[tid][0] + b_gate[tid]);
}

// ---------------- 5) chunk pooled mean -> mom, dec ----------------

__global__ void pool_kernel(const float* __restrict__ xs, const float* __restrict__ W_mom,
                            const float* __restrict__ b_mom, const float* __restrict__ W_dec,
                            const float* __restrict__ b_dec, float* __restrict__ mom,
                            float* __restrict__ dec) {
  __shared__ float red[4][256];
  int tid = threadIdx.x;
  int bidx = blockIdx.x / NC_, c = blockIdx.x % NC_;
  size_t row0 = (size_t)bidx * N_ + c * CH_;

  float pm[4] = {0, 0, 0, 0}, pd[4] = {0, 0, 0, 0};
  for (int d = tid; d < DIM_; d += 256) {
    float s = 0.f;
    for (int t = 0; t < CH_; ++t) s += xs[(row0 + t) * DIM_ + d];
    s *= (1.f / (float)CH_);
#pragma unroll
    for (int h = 0; h < H_; ++h) { pm[h] += s * W_mom[d * H_ + h]; pd[h] += s * W_dec[d * H_ + h]; }
  }
#pragma unroll
  for (int h = 0; h < H_; ++h) red[h][tid] = pm[h];
  __syncthreads();
  for (int o = 128; o > 0; o >>= 1) {
    if (tid < o) for (int h = 0; h < H_; ++h) red[h][tid] += red[h][tid + o];
    __syncthreads();
  }
  if (tid < H_) mom[((size_t)(bidx * H_ + tid)) * NC_ + c] = sigmoidf_(red[tid][0] + b_mom[tid]);
  __syncthreads();
#pragma unroll
  for (int h = 0; h < H_; ++h) red[h][tid] = pd[h];
  __syncthreads();
  for (int o = 128; o > 0; o >>= 1) {
    if (tid < o) for (int h = 0; h < H_; ++h) red[h][tid] += red[h][tid + o];
    __syncthreads();
  }
  if (tid < H_) dec[((size_t)(bidx * H_ + tid)) * NC_ + c] = sigmoidf_(red[tid][0] + b_dec[tid]);
}

// ---------------- 6) weight prep: bf16 + transposed layouts so B fragments are K-contiguous ----
// W1t[h][hid][dh], mw1t same layout f32 ; W2t[h][dh][hid], mw2t same layout f32 ; W2b direct bf16.

__global__ void prep_kernel(const float* __restrict__ mw1, const float* __restrict__ mw2,
                            __bf16* __restrict__ W1t, float* __restrict__ mw1t,
                            __bf16* __restrict__ W2t, float* __restrict__ mw2t,
                            __bf16* __restrict__ W2b) {
  int idx = blockIdx.x * 256 + threadIdx.x;      // [0, H*WEL_)
  int h = idx / WEL_;
  int e = idx % WEL_;
  int n1 = e / DH_, k1 = e % DH_;                 // [hid][dh] <- mw1[dh][hid]
  float v1 = mw1[(size_t)h * WEL_ + (size_t)k1 * HID_ + n1];
  W1t[idx] = (__bf16)v1;
  mw1t[idx] = v1;
  int n2 = e / HID_, k2 = e % HID_;               // [dh][hid] <- mw2[hid][dh]
  float v2 = mw2[(size_t)h * WEL_ + (size_t)k2 * DH_ + n2];
  W2t[idx] = (__bf16)v2;
  mw2t[idx] = v2;
  W2b[idx] = (__bf16)mw2[idx];                    // [hid][dh] direct
}

// ---------------- 7) per-(b,h,chunk) memory-MLP gradient (WMMA fwd + analytic bwd) ----------------

__global__ void grad_kernel(const float* __restrict__ kbuf, const float* __restrict__ vbuf,
                            const float* __restrict__ lr, const __bf16* __restrict__ W1t,
                            const __bf16* __restrict__ W2t, const __bf16* __restrict__ W2b,
                            const float* __restrict__ mg, const float* __restrict__ mb,
                            __bf16* __restrict__ gw1t, __bf16* __restrict__ gw2t,
                            float* __restrict__ gg, float* __restrict__ gb) {
  __shared__ __bf16 Kc[CH_ * DH_];     // 16 KB [t][d]
  __shared__ __bf16 Abuf[CH_ * HID_];  // 64 KB gelu acts [t][hid]; later dh1 transposed [hid][t]
  __shared__ float  Y[CH_ * DH_];      // 32 KB y0
  __shared__ __bf16 DY[CH_ * DH_];     // 16 KB dy0 [t][d]
  __shared__ __bf16 DYt[DH_ * CH_];    // 16 KB dy0 [d][t]
  __shared__ float  dgdb[2 * DH_];     // 1 KB

  int tid = threadIdx.x;
  int wave = tid >> 5;
  int lane = tid & 31;
  int half = lane >> 4, col = lane & 15;

  int cid = blockIdx.x;                 // (b*H + h)*NC + c
  int c = cid % NC_;
  int bh = cid / NC_;
  int h = bh % H_, bidx = bh / H_;
  size_t row0 = (size_t)bidx * N_ + c * CH_;

  const __bf16* W1 = W1t + (size_t)h * WEL_;   // [hid][dh]
  const __bf16* W2 = W2t + (size_t)h * WEL_;   // [dh][hid]
  const __bf16* W2r = W2b + (size_t)h * WEL_;  // [hid][dh]
  const float* G  = mg + h * DH_;
  const float* Bb = mb + h * DH_;

  for (int i = tid; i < CH_ * (DH_ / 4); i += 256) {
    int t = i >> 5, d4 = (i & 31) * 4;
    float4 f = *(const float4*)(kbuf + (row0 + t) * DIM_ + h * DH_ + d4);
    Kc[t * DH_ + d4 + 0] = (__bf16)f.x;
    Kc[t * DH_ + d4 + 1] = (__bf16)f.y;
    Kc[t * DH_ + d4 + 2] = (__bf16)f.z;
    Kc[t * DH_ + d4 + 3] = (__bf16)f.w;
  }
  if (tid < 2 * DH_) dgdb[tid] = 0.f;
  __syncthreads();

  // ---- fwd GEMM1: h1 = Kc @ W1, gelu -> Abuf. 4x32 tiles.
  for (int tile = wave; tile < 128; tile += 8) {
    int mt = tile >> 5, nt = tile & 31;
    v8f acc = zero8();
    for (int kk = 0; kk < DH_; kk += 32) {
      v16bf a = load_afrag(Kc + mt * 16 * DH_ + kk, DH_, 1);
      v16bf b = load_bfrag(W1 + (size_t)(nt * 16) * DH_ + kk, 1, DH_);
      acc = wmma_bf16(a, b, acc);
    }
#pragma unroll
    for (int r = 0; r < 8; ++r)
      Abuf[(mt * 16 + r + 8 * half) * HID_ + nt * 16 + col] = (__bf16)geluf(acc[r]);
  }
  __syncthreads();

  // ---- fwd GEMM2: Y = Kc + Abuf @ W2. 4x8 tiles.
  for (int tile = wave; tile < 32; tile += 8) {
    int mt = tile >> 3, nt = tile & 7;
    v8f acc = zero8();
    for (int kk = 0; kk < HID_; kk += 32) {
      v16bf a = load_afrag(Abuf + mt * 16 * HID_ + kk, HID_, 1);
      v16bf b = load_bfrag(W2 + (size_t)(nt * 16) * HID_ + kk, 1, HID_);
      acc = wmma_bf16(a, b, acc);
    }
#pragma unroll
    for (int r = 0; r < 8; ++r) {
      int mm = mt * 16 + r + 8 * half, nn = nt * 16 + col;
      Y[mm * DH_ + nn] = acc[r] + (float)Kc[mm * DH_ + nn];
    }
  }
  __syncthreads();

  // ---- LayerNorm fwd + dpred + LN backward -> DY, DYt (thread per token)
  if (tid < CH_) {
    int t = tid;
    const float* vp = vbuf + (row0 + t) * DIM_ + h * DH_;
    float lrt = lr[(size_t)bh * N_ + c * CH_ + t];
    float mu = 0.f;
    for (int d = 0; d < DH_; ++d) mu += Y[t * DH_ + d];
    mu *= (1.f / DH_);
    float var = 0.f;
    for (int d = 0; d < DH_; ++d) { float z = Y[t * DH_ + d] - mu; var += z * z; }
    var *= (1.f / DH_);
    float rstd = rsqrtf(var + EPS_);
    float coef = 2.f * lrt / (float)DH_;
    float s1 = 0.f, s2 = 0.f;
    for (int d = 0; d < DH_; ++d) {
      float yh = (Y[t * DH_ + d] - mu) * rstd;
      float dp = coef * ((yh * G[d] + Bb[d]) - vp[d]);
      atomicAdd(&dgdb[d], dp * yh);
      atomicAdd(&dgdb[DH_ + d], dp);
      float dyh = dp * G[d];
      s1 += dyh; s2 += dyh * yh;
    }
    s1 *= (1.f / DH_); s2 *= (1.f / DH_);
    for (int d = 0; d < DH_; ++d) {
      float yh = (Y[t * DH_ + d] - mu) * rstd;
      float dp = coef * ((yh * G[d] + Bb[d]) - vp[d]);
      float dyh = dp * G[d];
      __bf16 dv = (__bf16)(rstd * (dyh - s1 - yh * s2));
      DY[t * DH_ + d] = dv;
      DYt[d * CH_ + t] = dv;
    }
  }
  __syncthreads();
  if (tid < DH_) {
    gg[(size_t)cid * DH_ + tid] = dgdb[tid];
    gb[(size_t)cid * DH_ + tid] = dgdb[DH_ + tid];
  }

  // ---- dW2 = Abuf^T @ DY  [hid=512, dh=128]: 32x8 tiles, K=64. Stored transposed [dh][hid].
  for (int tile = wave; tile < 256; tile += 8) {
    int mt = tile >> 3, nt = tile & 7;
    v8f acc = zero8();
    for (int kk = 0; kk < CH_; kk += 32) {
      v16bf a = load_afrag(Abuf + mt * 16 + kk * HID_, 1, HID_);   // A^T (strided)
      v16bf b = load_bfrag(DYt + (size_t)(nt * 16) * CH_ + kk, 1, CH_);
      acc = wmma_bf16(a, b, acc);
    }
    store_cfrag(gw2t + (size_t)cid * WEL_ + (size_t)(nt * 16) * HID_ + mt * 16, 1, HID_, acc);
  }
  __syncthreads();

  // ---- da = DY @ W2^T (+ h1 recompute); dh1 = da * gelu'(h1) -> Abuf transposed [hid][t].
  for (int tile = wave; tile < 128; tile += 8) {
    int mt = tile >> 5, nt = tile & 31;
    v8f da = zero8();
    v8f h1 = zero8();
    for (int kk = 0; kk < DH_; kk += 32) {
      v16bf a = load_afrag(DY + mt * 16 * DH_ + kk, DH_, 1);
      v16bf b = load_bfrag(W2r + (size_t)(nt * 16) * DH_ + kk, 1, DH_);  // W2^T, K-contig
      da = wmma_bf16(a, b, da);
      v16bf a2 = load_afrag(Kc + mt * 16 * DH_ + kk, DH_, 1);
      v16bf b2 = load_bfrag(W1 + (size_t)(nt * 16) * DH_ + kk, 1, DH_);
      h1 = wmma_bf16(a2, b2, h1);
    }
#pragma unroll
    for (int r = 0; r < 8; ++r) {
      int mm = mt * 16 + r + 8 * half, nn = nt * 16 + col;
      Abuf[nn * CH_ + mm] = (__bf16)(da[r] * gelu_grad(h1[r]));    // transposed store
    }
  }
  __syncthreads();

  // ---- dW1 = Kc^T @ dh1  [dh=128, hid=512]: 8x32 tiles, K=64. Stored transposed [hid][dh].
  for (int tile = wave; tile < 256; tile += 8) {
    int mt = tile >> 5, nt = tile & 31;
    v8f acc = zero8();
    for (int kk = 0; kk < CH_; kk += 32) {
      v16bf a = load_afrag(Kc + mt * 16 + kk * DH_, 1, DH_);            // Kc^T (strided)
      v16bf b = load_bfrag(Abuf + (size_t)(nt * 16) * CH_ + kk, 1, CH_); // dh1t, K-contig
      acc = wmma_bf16(a, b, acc);
    }
    store_cfrag(gw1t + (size_t)cid * WEL_ + (size_t)(nt * 16) * DH_ + mt * 16, 1, DH_, acc);
  }
}

// ---------------- 8) gated scans (element-parallel, sequential over NC) ----------------

template <typename T>
__global__ void scan_kernel(const T* __restrict__ grads, const float* __restrict__ w0,
                            const float* __restrict__ mom, const float* __restrict__ dec,
                            T* __restrict__ wout, int E) {
  int e = blockIdx.x * blockDim.x + threadIdx.x;
  int bh = blockIdx.y;
  int h = bh % H_;
  if (e >= E) return;
  float m = 0.f;
  float w = w0[(size_t)h * E + e];
  for (int t = 0; t < NC_; ++t) {
    size_t idx = ((size_t)(bh * NC_ + t)) * E + e;
    wout[idx] = (T)w;
    float mm = mom[bh * NC_ + t];
    float dd = dec[bh * NC_ + t];
    m = mm * m - (float)grads[idx];
    w = (1.f - dd) * w + m;
  }
}

// ---------------- 9) retrieval: mem_model(W_prev, q) * gate -> O ----------------

__global__ void retrieve_kernel(const float* __restrict__ qbuf, const __bf16* __restrict__ pw1t,
                                const __bf16* __restrict__ pw2t, const float* __restrict__ pg,
                                const float* __restrict__ pb, const float* __restrict__ gate,
                                float* __restrict__ O) {
  __shared__ __bf16 Qc[CH_ * DH_];
  __shared__ __bf16 Abuf[CH_ * HID_];
  __shared__ float  Y[CH_ * DH_];

  int tid = threadIdx.x;
  int wave = tid >> 5;
  int lane = tid & 31;
  int half = lane >> 4, col = lane & 15;

  int cid = blockIdx.x;
  int c = cid % NC_;
  int bh = cid / NC_;
  int h = bh % H_, bidx = bh / H_;
  size_t row0 = (size_t)bidx * N_ + c * CH_;
  (void)h;

  const __bf16* W1 = pw1t + (size_t)cid * WEL_;   // [hid][dh]
  const __bf16* W2 = pw2t + (size_t)cid * WEL_;   // [dh][hid]
  const float* G  = pg + (size_t)cid * DH_;
  const float* Bb = pb + (size_t)cid * DH_;

  for (int i = tid; i < CH_ * (DH_ / 4); i += 256) {
    int t = i >> 5, d4 = (i & 31) * 4;
    float4 f = *(const float4*)(qbuf + (row0 + t) * DIM_ + (cid / NC_ % H_) * DH_ + d4);
    Qc[t * DH_ + d4 + 0] = (__bf16)f.x;
    Qc[t * DH_ + d4 + 1] = (__bf16)f.y;
    Qc[t * DH_ + d4 + 2] = (__bf16)f.z;
    Qc[t * DH_ + d4 + 3] = (__bf16)f.w;
  }
  __syncthreads();

  for (int tile = wave; tile < 128; tile += 8) {
    int mt = tile >> 5, nt = tile & 31;
    v8f acc = zero8();
    for (int kk = 0; kk < DH_; kk += 32) {
      v16bf a = load_afrag(Qc + mt * 16 * DH_ + kk, DH_, 1);
      v16bf b = load_bfrag(W1 + (size_t)(nt * 16) * DH_ + kk, 1, DH_);
      acc = wmma_bf16(a, b, acc);
    }
#pragma unroll
    for (int r = 0; r < 8; ++r)
      Abuf[(mt * 16 + r + 8 * half) * HID_ + nt * 16 + col] = (__bf16)geluf(acc[r]);
  }
  __syncthreads();

  for (int tile = wave; tile < 32; tile += 8) {
    int mt = tile >> 3, nt = tile & 7;
    v8f acc = zero8();
    for (int kk = 0; kk < HID_; kk += 32) {
      v16bf a = load_afrag(Abuf + mt * 16 * HID_ + kk, HID_, 1);
      v16bf b = load_bfrag(W2 + (size_t)(nt * 16) * HID_ + kk, 1, HID_);
      acc = wmma_bf16(a, b, acc);
    }
#pragma unroll
    for (int r = 0; r < 8; ++r) {
      int mm = mt * 16 + r + 8 * half, nn = nt * 16 + col;
      Y[mm * DH_ + nn] = acc[r] + (float)Qc[mm * DH_ + nn];
    }
  }
  __syncthreads();

  if (tid < CH_) {
    int t = tid;
    float gv = gate[(size_t)bh * N_ + c * CH_ + t];
    float mu = 0.f;
    for (int d = 0; d < DH_; ++d) mu += Y[t * DH_ + d];
    mu *= (1.f / DH_);
    float var = 0.f;
    for (int d = 0; d < DH_; ++d) { float z = Y[t * DH_ + d] - mu; var += z * z; }
    var *= (1.f / DH_);
    float rstd = rsqrtf(var + EPS_);
    int hh = bh % H_;
    for (int d = 0; d < DH_; ++d) {
      float yh = (Y[t * DH_ + d] - mu) * rstd;
      O[(row0 + t) * DIM_ + hh * DH_ + d] = (yh * G[d] + Bb[d]) * gv;
    }
  }
}

// ---------------- host: launch pipeline ----------------

extern "C" void kernel_launch(void* const* d_in, const int* in_sizes, int n_in,
                              void* d_out, int out_size, void* d_ws, size_t ws_size,
                              hipStream_t stream) {
  const float* x      = (const float*)d_in[0];
  const float* g_ret  = (const float*)d_in[1];
  const float* g_sto  = (const float*)d_in[2];
  const float* Wq     = (const float*)d_in[3];
  const float* Wk     = (const float*)d_in[4];
  const float* Wv     = (const float*)d_in[5];
  const float* W_lr   = (const float*)d_in[6];
  const float* b_lr   = (const float*)d_in[7];
  const float* W_mom  = (const float*)d_in[8];
  const float* b_mom  = (const float*)d_in[9];
  const float* W_dec  = (const float*)d_in[10];
  const float* b_dec  = (const float*)d_in[11];
  const float* W_gate = (const float*)d_in[12];
  const float* b_gate = (const float*)d_in[13];
  const float* W_comb = (const float*)d_in[14];
  const float* b_comb = (const float*)d_in[15];
  const float* mw1    = (const float*)d_in[16];
  const float* mw2    = (const float*)d_in[17];
  const float* mg     = (const float*)d_in[18];
  const float* mb     = (const float*)d_in[19];
  float* out = (float*)d_out;

  const size_t ROWS = (size_t)B_ * N_;          // 4096
  const size_t MAT  = ROWS * DIM_;              // 2,097,152
  const size_t BHN  = (size_t)B_ * H_ * N_;     // 16,384
  const size_t CHK  = (size_t)B_ * H_ * NC_;    // 256
  const size_t WEL  = (size_t)WEL_;             // 65,536

  float* wsf = (float*)d_ws;
  size_t off = 0;
  auto alloc = [&](size_t n) { float* p = wsf + off; off += n; return p; };
  float* xs    = alloc(MAT);
  float* xr    = alloc(MAT);
  float* qbuf  = alloc(MAT);
  float* kbuf  = alloc(MAT);
  float* vbuf  = alloc(MAT);
  float* lr    = alloc(BHN);
  float* gate  = alloc(BHN);
  float* mom   = alloc(CHK);
  float* dec   = alloc(CHK);
  float* gg    = alloc(CHK * DH_);
  float* gb    = alloc(CHK * DH_);
  float* wg    = alloc(CHK * DH_);
  float* wb    = alloc(CHK * DH_);
  float* mw1t  = alloc(H_ * WEL);
  float* mw2t  = alloc(H_ * WEL);
  float* O     = alloc(MAT);
  __bf16* bptr = (__bf16*)(wsf + off);
  __bf16* W1t  = bptr; bptr += H_ * WEL;
  __bf16* W2t  = bptr; bptr += H_ * WEL;
  __bf16* W2b  = bptr; bptr += H_ * WEL;
  __bf16* gw1t = bptr; bptr += CHK * WEL;
  __bf16* gw2t = bptr; bptr += CHK * WEL;
  __bf16* pw1t = bptr; bptr += CHK * WEL;
  __bf16* pw2t = bptr; bptr += CHK * WEL;

  // 1) pre-norms
  rms_kernel<<<(int)ROWS, 256, 0, stream>>>(x, g_ret, g_sto, xs, xr);

  // 2) projections (WMMA GEMM)
  dim3 gemm_grid(DIM_ / 64, (int)(ROWS / 64));
  gemm_kernel<<<gemm_grid, 256, 0, stream>>>(xr, Wq, nullptr, qbuf, (int)ROWS, DIM_, DIM_);
  gemm_kernel<<<gemm_grid, 256, 0, stream>>>(xs, Wk, nullptr, kbuf, (int)ROWS, DIM_, DIM_);
  gemm_kernel<<<gemm_grid, 256, 0, stream>>>(xs, Wv, nullptr, vbuf, (int)ROWS, DIM_, DIM_);

  // 3) l2 normalize q, k per (token, head)
  l2n_kernel<<<(int)(ROWS * H_), 128, 0, stream>>>(qbuf);
  l2n_kernel<<<(int)(ROWS * H_), 128, 0, stream>>>(kbuf);

  // 4) lr & gate
  aux_kernel<<<(int)ROWS, 256, 0, stream>>>(xs, xr, W_lr, b_lr, W_gate, b_gate, lr, gate);

  // 5) pooled -> mom, dec
  pool_kernel<<<B_ * NC_, 256, 0, stream>>>(xs, W_mom, b_mom, W_dec, b_dec, mom, dec);

  // 6) weight prep (bf16 + transposes)
  prep_kernel<<<(int)(H_ * WEL / 256), 256, 0, stream>>>(mw1, mw2, W1t, mw1t, W2t, mw2t, W2b);

  // 7) per-chunk grads (WMMA fwd+bwd)
  grad_kernel<<<(int)CHK, 256, 0, stream>>>(kbuf, vbuf, lr, W1t, W2t, W2b, mg, mb,
                                            gw1t, gw2t, gg, gb);

  // 8) momentum + decay scans
  dim3 scan_grid((int)(WEL / 256), B_ * H_);
  scan_kernel<__bf16><<<scan_grid, 256, 0, stream>>>(gw1t, mw1t, mom, dec, pw1t, (int)WEL);
  scan_kernel<__bf16><<<scan_grid, 256, 0, stream>>>(gw2t, mw2t, mom, dec, pw2t, (int)WEL);
  dim3 scan_grid2(1, B_ * H_);
  scan_kernel<float><<<scan_grid2, 128, 0, stream>>>(gg, mg, mom, dec, wg, DH_);
  scan_kernel<float><<<scan_grid2, 128, 0, stream>>>(gb, mb, mom, dec, wb, DH_);

  // 9) retrieval (WMMA fwd) * gate
  retrieve_kernel<<<(int)CHK, 256, 0, stream>>>(qbuf, pw1t, pw2t, wg, wb, gate, O);

  // 10) combine: out = O @ W_comb + b_comb
  gemm_kernel<<<gemm_grid, 256, 0, stream>>>(O, W_comb, b_comb, out, (int)ROWS, DIM_, DIM_);
}
